// MultiHeadAttention_26525718020147
// MI455X (gfx1250) — matrix operand order repaired and checked by reading, not compiled, not run on hardware
//
#include <hip/hip_runtime.h>
#include <hip/hip_bf16.h>

typedef __attribute__((ext_vector_type(16))) _Float16 v16h;
typedef __attribute__((ext_vector_type(8)))  _Float16 v8h;
typedef __attribute__((ext_vector_type(8)))  float    v8f;
typedef __attribute__((ext_vector_type(4)))  float    v4f;
typedef __attribute__((ext_vector_type(4)))  unsigned int v4u;
typedef __attribute__((ext_vector_type(4)))  int      v4i;
typedef __attribute__((ext_vector_type(8)))  int      v8i;

union V16 { v16h v; v8h h[2]; };

#define EMBED 1024
#define SEQ   2048
#define NHEAD 16
#define HDIM  64

// ---------------------------------------------------------------------------
// TDM: 2D tile load Global -> LDS (D# built per CDNA5 ISA ch.8).
// dim0/tile0 = elements along a line, stride0 = line stride (elements),
// dim1/tile1 = number of lines. data_size = 2 bytes (f16).
// padi/pada: LDS padding codes (pad_interval: 1<<n 8B-units per row,
// pad_amount: (n+1) DWORDs inserted) -> keeps LDS rows 16B-aligned,
// bank-conflict free.
// ---------------------------------------------------------------------------
__device__ __forceinline__ void tdm_load_2d_f16(
    unsigned int lds_off, const void* gptr,
    unsigned int dim0, unsigned int dim1,
    unsigned int til0, unsigned int til1,
    unsigned int stride0, unsigned int padi, unsigned int pada)
{
  unsigned long long ga = (unsigned long long)(uintptr_t)gptr;
  v4u g0;
  g0[0] = 1u;                                            // count=1 (valid D#)
  g0[1] = lds_off;                                       // lds_addr
  g0[2] = (unsigned int)ga;                              // global_addr lo
  g0[3] = (unsigned int)((ga >> 32) & 0x01FFFFFFu) | (2u << 30); // hi + type=2
  v8i g1;
  g1[0] = (int)((1u << 16) |                             // data_size = 2B
                (1u << 20) |                             // pad_enable
                (padi << 22) | (pada << 25));            // pad interval/amount
  g1[1] = (int)((dim0 & 0xFFFFu) << 16);                 // tensor_dim0 lo16
  g1[2] = (int)((dim0 >> 16) | ((dim1 & 0xFFFFu) << 16));// dim0 hi / dim1 lo
  g1[3] = (int)((dim1 >> 16) | (til0 << 16));            // dim1 hi / tile_dim0
  g1[4] = (int)til1;                                     // tile_dim1 (dim2=0)
  g1[5] = (int)stride0;                                  // tensor_dim0_stride
  g1[6] = 0;
  g1[7] = 0;
  v4i z4 = {};
#if defined(__clang_major__) && (__clang_major__ >= 23)
  v8i z8 = {};
  __builtin_amdgcn_tensor_load_to_lds(g0, g1, z4, z4, z8, 0);
#else
  __builtin_amdgcn_tensor_load_to_lds(g0, g1, z4, z4, 0);
#endif
}

// ---------------------------------------------------------------------------
// Generic projection GEMM: Y = X(fp32)[M,1024] @ W(fp32)[1024,1024] + bias
// MODE 0: fp32 row-major [M,1024]          (final output projection)
// MODE 1: f16  [B,H,S,D]                   (Q, K)
// MODE 2: f16  [B,H,D,S]  (transposed)     (V -> contiguous B-operand later)
// ---------------------------------------------------------------------------
template<int MODE>
__global__ __launch_bounds__(256) void gemm_proj(
    const float* __restrict__ X, const float* __restrict__ W,
    const float* __restrict__ bias, void* __restrict__ Yout)
{
  __shared__ alignas(16) _Float16 Xs[128][40];   // [row][k], padded stride
  __shared__ alignas(16) _Float16 Ws[128][40];   // [col][k] (W transposed)

  const int tid     = threadIdx.x;
  const int lane    = tid & 31;
  const int w       = tid >> 5;
  const int l15     = lane & 15;
  const int halfsel = lane >> 4;

  const int gm0 = blockIdx.x * 128;
  const int gn0 = blockIdx.y * 128;
  const int wm  = (w & 3) * 32;     // wave row origin in tile
  const int wn  = (w >> 2) * 64;    // wave col origin in tile

  const int kb  = halfsel * 8;      // A-fragment K base (ISA A layout)
  const int kbb = halfsel * 16;     // unused; B fragments also use kb

  v8f acc[2][4];
  #pragma unroll
  for (int nt = 0; nt < 4; ++nt) {
    float bv = bias[gn0 + wn + nt * 16 + l15];
    #pragma unroll
    for (int mt = 0; mt < 2; ++mt)
      #pragma unroll
      for (int r = 0; r < 8; ++r)
        acc[mt][nt][r] = bv;
  }

  const int xrow = tid >> 1;          // 0..127
  const int xcol = (tid & 1) * 16;    // 0 or 16
  const int wk   = tid >> 3;          // 0..31
  const int wn0  = (tid & 7) * 16;    // 0..112

  for (int k0 = 0; k0 < EMBED; k0 += 32) {
    // ---- stage X tile (128x32) fp32 -> f16, row-major ----
    const float* xp = X + (size_t)(gm0 + xrow) * EMBED + k0 + xcol;
    v4f x0 = *(const v4f*)(xp + 0);
    v4f x1 = *(const v4f*)(xp + 4);
    v4f x2 = *(const v4f*)(xp + 8);
    v4f x3 = *(const v4f*)(xp + 12);
    v8h hx0, hx1;
    #pragma unroll
    for (int j = 0; j < 4; ++j) {
      hx0[j]     = (_Float16)x0[j];  hx0[4 + j] = (_Float16)x1[j];
      hx1[j]     = (_Float16)x2[j];  hx1[4 + j] = (_Float16)x3[j];
    }
    *(v8h*)&Xs[xrow][xcol]     = hx0;
    *(v8h*)&Xs[xrow][xcol + 8] = hx1;

    // ---- stage W tile (32x128) fp32 -> f16, transposed into Ws[col][k] ----
    const float* wp = W + (size_t)(k0 + wk) * EMBED + gn0 + wn0;
    #pragma unroll
    for (int j = 0; j < 16; j += 4) {
      v4f wv = *(const v4f*)(wp + j);
      Ws[wn0 + j + 0][wk] = (_Float16)wv[0];
      Ws[wn0 + j + 1][wk] = (_Float16)wv[1];
      Ws[wn0 + j + 2][wk] = (_Float16)wv[2];
      Ws[wn0 + j + 3][wk] = (_Float16)wv[3];
    }
    // prefetch next K-step tiles into cache
    if (k0 + 32 < EMBED) {
      __builtin_prefetch(xp + 32, 0, 1);
      __builtin_prefetch(wp + 32 * EMBED, 0, 1);
    }
    __syncthreads();

    // ---- fragments + 8 WMMAs ----
    V16 af[2];
    #pragma unroll
    for (int mt = 0; mt < 2; ++mt) {
      int row = wm + mt * 16 + l15;
      af[mt].h[0] = *(const v8h*)&Xs[row][kb];        // K = kb..kb+7
      af[mt].h[1] = *(const v8h*)&Xs[row][kb + 16];   // K = kb+16..kb+23
    }
    #pragma unroll
    for (int nt = 0; nt < 4; ++nt) {
      int col = wn + nt * 16 + l15;
      V16 bf;
      bf.h[0] = *(const v8h*)&Ws[col][kb];            // K = kb..kb+7
      bf.h[1] = *(const v8h*)&Ws[col][kb + 16];       // K = kb+16..kb+23
      #pragma unroll
      for (int mt = 0; mt < 2; ++mt)
        acc[mt][nt] = __builtin_amdgcn_wmma_f32_16x16x32_f16(
            false, af[mt].v, false, bf.v, (short)0, acc[mt][nt], false, false);
    }
    __syncthreads();
  }

  // ---- epilogue ----
  #pragma unroll
  for (int mt = 0; mt < 2; ++mt) {
    #pragma unroll
    for (int nt = 0; nt < 4; ++nt) {
      #pragma unroll
      for (int r = 0; r < 8; ++r) {
        int gm = gm0 + wm + mt * 16 + r + halfsel * 8;  // C layout row
        int gn = gn0 + wn + nt * 16 + l15;              // C layout col
        float vv = acc[mt][nt][r];
        if (MODE == 0) {
          ((float*)Yout)[(size_t)gm * EMBED + gn] = vv;
        } else {
          int b = gm >> 11, s = gm & 2047;
          int h = gn >> 6,  d = gn & 63;
          if (MODE == 1)
            ((_Float16*)Yout)[((size_t)(b * NHEAD + h) * SEQ + s) * HDIM + d] =
                (_Float16)vv;
          else
            ((_Float16*)Yout)[((size_t)(b * NHEAD + h) * HDIM + d) * SEQ + s] =
                (_Float16)vv;
        }
      }
    }
  }
}

// ---------------------------------------------------------------------------
// Flash-style fused attention with TDM double-buffered K/V staging in LDS.
// One wave = 16 query rows; workgroup = 128 rows; K/V 32-key tiles are DMA'd
// once per workgroup (wave 0 issues, TENSORcnt-tracked) and shared by all 8
// waves. Q,K: f16 [B,H,S,D]; V: f16 [B,H,D,S]; out fp32 [B,S,E].
// TDM pads LDS rows: Ks row = 64 f16 + 16B pad = 144B; Vs row = 32 f16 + 16B
// pad = 80B (16B-aligned v8h reads, conflict-free bank strides).
// ---------------------------------------------------------------------------
__global__ __launch_bounds__(256) void attn_kernel(
    const _Float16* __restrict__ Q, const _Float16* __restrict__ K,
    const _Float16* __restrict__ Vt, float* __restrict__ ctx)
{
  __shared__ alignas(16) _Float16 Ks[2][32][72];   // [buf][key][d + pad]
  __shared__ alignas(16) _Float16 Vs[2][64][40];   // [buf][d][key + pad]
  __shared__ alignas(16) _Float16 Ps[8][16][40];   // per-wave P bounce

  const int tid     = threadIdx.x;
  const int lane    = tid & 31;
  const int w       = tid >> 5;
  const int l15     = lane & 15;
  const int halfsel = lane >> 4;

  const int bh = blockIdx.y;                 // b*16 + h
  const int q0 = blockIdx.x * 128 + w * 16;  // wave's query-row base

  const _Float16* Qb = Q  + (size_t)bh * SEQ * HDIM;
  const _Float16* Kb = K  + (size_t)bh * SEQ * HDIM;
  const _Float16* Vb = Vt + (size_t)bh * HDIM * SEQ;

  const int kb  = halfsel * 8;
  const int kbb = halfsel * 16;

  // Q A-fragments (loop-invariant): d in [0,32) and [32,64)
  V16 qf[2];
  #pragma unroll
  for (int c = 0; c < 2; ++c) {
    const _Float16* qp = Qb + (size_t)(q0 + l15) * HDIM + c * 32;
    qf[c].h[0] = *(const v8h*)(qp + kb);
    qf[c].h[1] = *(const v8h*)(qp + kb + 16);
  }

  v8f   o[4];
  float m[8], l[8];
  #pragma unroll
  for (int nt = 0; nt < 4; ++nt) o[nt] = (v8f){};
  #pragma unroll
  for (int r = 0; r < 8; ++r) { m[r] = -1e30f; l[r] = 0.0f; }

  const float sc = 0.125f * 1.44269504088896340736f;  // 1/sqrt(64) * log2(e)

  // Prime the pipeline: DMA K/V tile 0 into buffer 0 (wave 0 only).
  if (w == 0) {
    // K tile: 32 lines of 64 f16, line stride 64; pad 4 DW per 128B row.
    tdm_load_2d_f16((unsigned int)(uintptr_t)&Ks[0][0][0], Kb,
                    HDIM, SEQ, HDIM, 32, HDIM, /*padi=*/4, /*pada=*/3);
    // V tile: 64 lines of 32 f16, line stride 2048; pad 4 DW per 64B row.
    tdm_load_2d_f16((unsigned int)(uintptr_t)&Vs[0][0][0], Vb,
                    SEQ, HDIM, 32, HDIM, SEQ, /*padi=*/3, /*pada=*/3);
  }

  for (int kblk = 0; kblk < SEQ; kblk += 32) {
    const int buf = (kblk >> 5) & 1;
    __syncthreads();   // everyone done reading buf^1 from previous iteration
    if (w == 0) {
      if (kblk + 32 < SEQ) {
        tdm_load_2d_f16((unsigned int)(uintptr_t)&Ks[buf ^ 1][0][0],
                        Kb + (size_t)(kblk + 32) * HDIM,
                        HDIM, SEQ, HDIM, 32, HDIM, 4, 3);
        tdm_load_2d_f16((unsigned int)(uintptr_t)&Vs[buf ^ 1][0][0],
                        Vb + (kblk + 32),
                        SEQ, HDIM, 32, HDIM, SEQ, 3, 3);
        __builtin_amdgcn_s_wait_tensorcnt(2);  // current tile (oldest 2) done
      } else {
        __builtin_amdgcn_s_wait_tensorcnt(0);
      }
    }
    __syncthreads();   // buf's K/V visible to all waves

    // ---- scores: two 16x16 tiles, D=64 via two K=32 WMMAs each ----
    v8f s[2];
    #pragma unroll
    for (int t = 0; t < 2; ++t) {
      const _Float16* kp = &Ks[buf][t * 16 + l15][0];
      V16 bf0, bf1;
      bf0.h[0] = *(const v8h*)(kp + kb);
      bf0.h[1] = *(const v8h*)(kp + kb + 16);
      bf1.h[0] = *(const v8h*)(kp + 32 + kb);
      bf1.h[1] = *(const v8h*)(kp + 32 + kb + 16);
      v8f z = {};
      s[t] = __builtin_amdgcn_wmma_f32_16x16x32_f16(
          false, qf[0].v, false, bf0.v, (short)0, z, false, false);
      s[t] = __builtin_amdgcn_wmma_f32_16x16x32_f16(
          false, qf[1].v, false, bf1.v, (short)0, s[t], false, false);
    }

    // ---- online softmax (fp32, base-2) ----
    #pragma unroll
    for (int r = 0; r < 8; ++r) {
      s[0][r] *= sc;  s[1][r] *= sc;
      float mx = fmaxf(s[0][r], s[1][r]);
      mx = fmaxf(mx, __shfl_xor(mx, 1));
      mx = fmaxf(mx, __shfl_xor(mx, 2));
      mx = fmaxf(mx, __shfl_xor(mx, 4));
      mx = fmaxf(mx, __shfl_xor(mx, 8));
      float mnew  = fmaxf(m[r], mx);
      float alpha = exp2f(m[r] - mnew);
      m[r] = mnew;
      float p0 = exp2f(s[0][r] - mnew);
      float p1 = exp2f(s[1][r] - mnew);
      s[0][r] = p0;  s[1][r] = p1;
      float rs = p0 + p1;
      rs += __shfl_xor(rs, 1);
      rs += __shfl_xor(rs, 2);
      rs += __shfl_xor(rs, 4);
      rs += __shfl_xor(rs, 8);
      l[r] = l[r] * alpha + rs;
      #pragma unroll
      for (int nt = 0; nt < 4; ++nt) o[nt][r] *= alpha;
    }

    // ---- P: C-layout f32 -> A-layout f16 via per-wave LDS bounce ----
    #pragma unroll
    for (int r = 0; r < 8; ++r) {
      int row = r + halfsel * 8;
      Ps[w][row][l15]      = (_Float16)s[0][r];
      Ps[w][row][16 + l15] = (_Float16)s[1][r];
    }
    V16 pf;
    pf.h[0] = *(const v8h*)&Ps[w][l15][kb];
    pf.h[1] = *(const v8h*)&Ps[w][l15][kb + 16];

    // ---- O += P @ V ----
    #pragma unroll
    for (int nt = 0; nt < 4; ++nt) {
      const _Float16* vp = &Vs[buf][nt * 16 + l15][0];
      V16 vf;
      vf.h[0] = *(const v8h*)(vp + kb);
      vf.h[1] = *(const v8h*)(vp + kb + 16);
      o[nt] = __builtin_amdgcn_wmma_f32_16x16x32_f16(
          false, pf.v, false, vf.v, (short)0, o[nt], false, false);
    }
  }

  // ---- normalize + write ctx fp32 [B,S,E] ----
  const int b = bh >> 4, h = bh & 15;
  #pragma unroll
  for (int r = 0; r < 8; ++r) {
    float inv = 1.0f / l[r];
    int row = q0 + r + halfsel * 8;
    #pragma unroll
    for (int nt = 0; nt < 4; ++nt)
      ctx[(size_t)(b * SEQ + row) * EMBED + h * HDIM + nt * 16 + l15] =
          o[nt][r] * inv;
  }
}

// ---------------------------------------------------------------------------
extern "C" void kernel_launch(void* const* d_in, const int* in_sizes, int n_in,
                              void* d_out, int out_size, void* d_ws, size_t ws_size,
                              hipStream_t stream) {
  (void)in_sizes; (void)n_in; (void)out_size; (void)ws_size;
  const float* q  = (const float*)d_in[0];
  const float* k  = (const float*)d_in[1];
  const float* v  = (const float*)d_in[2];
  const float* Wq = (const float*)d_in[3];
  const float* bq = (const float*)d_in[4];
  const float* Wk = (const float*)d_in[5];
  const float* bk = (const float*)d_in[6];
  const float* Wv = (const float*)d_in[7];
  const float* bv = (const float*)d_in[8];
  const float* Wo = (const float*)d_in[9];
  const float* bo = (const float*)d_in[10];

  char* ws = (char*)d_ws;
  _Float16* Qh  = (_Float16*)(ws);                               // 16 MB
  _Float16* Kh  = (_Float16*)(ws + (size_t)16 * 1024 * 1024);    // 16 MB
  _Float16* Vth = (_Float16*)(ws + (size_t)32 * 1024 * 1024);    // 16 MB
  float*    ctx = (float*)   (ws + (size_t)48 * 1024 * 1024);    // 32 MB

  dim3 gg(64, 8), bb(256);
  hipLaunchKernelGGL((gemm_proj<1>), gg, bb, 0, stream, q, Wq, bq, (void*)Qh);
  hipLaunchKernelGGL((gemm_proj<1>), gg, bb, 0, stream, k, Wk, bk, (void*)Kh);
  hipLaunchKernelGGL((gemm_proj<2>), gg, bb, 0, stream, v, Wv, bv, (void*)Vth);
  hipLaunchKernelGGL(attn_kernel, dim3(16, 64), bb, 0, stream, Qh, Kh, Vth, ctx);
  hipLaunchKernelGGL((gemm_proj<0>), gg, bb, 0, stream, ctx, Wo, bo, d_out);
}
